// Dominant_50723563766558
// MI455X (gfx1250) — compile-verified, matrix operand order checked by reading, not run on hardware
//
#include <hip/hip_runtime.h>
#include <hip/hip_bf16.h>
#include <cstdint>

#define NNODES 50000

typedef __attribute__((ext_vector_type(16))) __bf16 v16bf;
typedef __attribute__((ext_vector_type(8)))  float  v8f;

static __device__ __forceinline__ __bf16 f2bf(float f) { return (__bf16)f; }

// ---------------------------------------------------------------------------
// GEMM + attention-score epilogue.
// H[N,OUT] = X[N,IN] @ W[OUT,IN]^T  (bf16 WMMA, fp32 accumulate)
// AS[N] = H @ a_src ; AD[N] = H @ a_dst  (fused, shfl reduction)
// One block = 256 threads = 8 waves, each wave owns a 16-row stripe.
// ---------------------------------------------------------------------------
template<int IN, int OUT>
__global__ __launch_bounds__(256)
void gemm_attn(const float* __restrict__ X, const float* __restrict__ W,
               const float* __restrict__ a_src, const float* __restrict__ a_dst,
               float* __restrict__ H, float* __restrict__ AS, float* __restrict__ AD,
               int nrows)
{
    constexpr int LDW = IN + 8;              // pad 16B -> avoid 64-bank conflicts
    __shared__ __bf16 sW[OUT * LDW];

    const int tid = threadIdx.x;
    for (int i = tid; i < OUT * IN; i += 256) {
        int r = i / IN, c = i % IN;
        sW[r * LDW + c] = f2bf(W[i]);
    }
    __syncthreads();

    const int wave  = tid >> 5;
    const int lane  = tid & 31;
    const int lrow  = lane & 15;             // row (A) / col (B) within tile
    const int khalf = (lane >> 4) * 8;       // K sub-offset per lane half
    const int rowbase = blockIdx.x * 128 + wave * 16;

    int arow = rowbase + lrow;
    int rrow = arow < nrows ? arow : nrows - 1;   // clamp loads, guard stores

    constexpr int NT = OUT / 16;
    v8f zero;
#pragma unroll
    for (int j = 0; j < 8; ++j) zero[j] = 0.0f;
    v8f c[NT];
#pragma unroll
    for (int t = 0; t < NT; ++t) c[t] = zero;

    for (int ks = 0; ks < IN; ks += 32) {
        // A fragment: 16x32 bf16 per ISA layout
        const float* xp = X + (size_t)rrow * IN + ks + khalf;
        const float4 f0 = *(const float4*)(xp);
        const float4 f1 = *(const float4*)(xp + 4);
        const float4 f2 = *(const float4*)(xp + 16);
        const float4 f3 = *(const float4*)(xp + 20);
        v16bf a;
        a[0]=f2bf(f0.x); a[1]=f2bf(f0.y); a[2]=f2bf(f0.z); a[3]=f2bf(f0.w);
        a[4]=f2bf(f1.x); a[5]=f2bf(f1.y); a[6]=f2bf(f1.z); a[7]=f2bf(f1.w);
        a[8]=f2bf(f2.x); a[9]=f2bf(f2.y); a[10]=f2bf(f2.z); a[11]=f2bf(f2.w);
        a[12]=f2bf(f3.x); a[13]=f2bf(f3.y); a[14]=f2bf(f3.z); a[15]=f2bf(f3.w);

#pragma unroll
        for (int nt = 0; nt < NT; ++nt) {
            // B fragment: B[k,n] = W[n,k] from LDS (symmetric lane layout)
            const __bf16* wp = &sW[(nt * 16 + lrow) * LDW + ks + khalf];
            v16bf b;
#pragma unroll
            for (int j = 0; j < 8; ++j) { b[j] = wp[j]; b[j + 8] = wp[16 + j]; }
            c[nt] = __builtin_amdgcn_wmma_f32_16x16x32_bf16(
                false, a, false, b, (short)0, c[nt], false, false);
        }
    }

    // Epilogue: store H, fuse alpha_s/alpha_d row dots.
    float ps[8], pd[8];
#pragma unroll
    for (int v = 0; v < 8; ++v) { ps[v] = 0.f; pd[v] = 0.f; }

    const int mhalf = (lane >> 4) * 8;       // D rows: v + 0 (lo lanes) / v + 8 (hi)
    const bool fullTile = (rowbase + 16) <= nrows;   // uniform per wave

    if (fullTile) {
        // fast path: no per-element guards -> straight-line b32 stores
#pragma unroll
        for (int nt = 0; nt < NT; ++nt) {
            int n = nt * 16 + lrow;
            float asv = a_src[n], adv = a_dst[n];
            float* hp = H + (size_t)(rowbase + mhalf) * OUT + n;
#pragma unroll
            for (int v = 0; v < 8; ++v) {
                float val = c[nt][v];
                hp[(size_t)v * OUT] = val;
                ps[v] += val * asv;
                pd[v] += val * adv;
            }
        }
    } else {
#pragma unroll
        for (int nt = 0; nt < NT; ++nt) {
            int n = nt * 16 + lrow;
            float asv = a_src[n], adv = a_dst[n];
#pragma unroll
            for (int v = 0; v < 8; ++v) {
                float val = c[nt][v];
                int m = rowbase + v + mhalf;
                if (m < nrows) H[(size_t)m * OUT + n] = val;
                ps[v] += val * asv;
                pd[v] += val * adv;
            }
        }
    }

#pragma unroll
    for (int v = 0; v < 8; ++v) {
        float s = ps[v], d = pd[v];
#pragma unroll
        for (int off = 1; off < 16; off <<= 1) {
            s += __shfl_xor(s, off, 32);
            d += __shfl_xor(d, off, 32);
        }
        int m = rowbase + v + mhalf;
        if (lrow == 0 && m < nrows) { AS[m] = s; AD[m] = d; }
    }
}

// ---------------------------------------------------------------------------
// Edge / softmax / scatter kernels
// ---------------------------------------------------------------------------
__global__ __launch_bounds__(256)
void build_edges(const long long* __restrict__ ei, int* __restrict__ src,
                 int* __restrict__ dst, int E, int Etot)
{
    int i = blockIdx.x * 256 + threadIdx.x;
    if (i >= Etot) return;
    if (i < E) { src[i] = (int)ei[i]; dst[i] = (int)ei[E + i]; }
    else       { src[i] = i - E;      dst[i] = i - E; }        // self loops
}

__global__ __launch_bounds__(256)
void fill_kernel(float* __restrict__ p, float v, int n)
{
    int i = blockIdx.x * 256 + threadIdx.x;
    if (i < n) p[i] = v;
}

static __device__ __forceinline__ float leaky(float e)
{
    return e > 0.f ? e : 0.2f * e;
}

static __device__ __forceinline__ void atomicMaxF(float* addr, float val)
{
    if (val >= 0.f) atomicMax((int*)addr, __float_as_int(val));
    else            atomicMin((unsigned int*)addr, __float_as_uint(val));
}

__global__ __launch_bounds__(256)
void edge_max(const int* __restrict__ src, const int* __restrict__ dst,
              const float* __restrict__ AS, const float* __restrict__ AD,
              float* __restrict__ M, int Etot)
{
    int i = blockIdx.x * 256 + threadIdx.x;
    if (i >= Etot) return;
    float e = leaky(AS[src[i]] + AD[dst[i]]);
    atomicMaxF(&M[dst[i]], e);
}

__global__ __launch_bounds__(256)
void edge_exp(const int* __restrict__ src, const int* __restrict__ dst,
              const float* __restrict__ AS, const float* __restrict__ AD,
              const float* __restrict__ M, float* __restrict__ Ssum,
              float* __restrict__ Wg, int Etot)
{
    int i = blockIdx.x * 256 + threadIdx.x;
    if (i >= Etot) return;
    int d = dst[i];
    float e = leaky(AS[src[i]] + AD[d]);
    float w = __expf(e - M[d]);
    Wg[i] = w;
    atomicAdd(&Ssum[d], w);
}

// one thread = 4 features of one edge; consecutive lanes -> contiguous features
template<int OUT>
__global__ __launch_bounds__(256)
void edge_scatter(const int* __restrict__ src, const int* __restrict__ dst,
                  const float* __restrict__ Wg, const float* __restrict__ Ssum,
                  const float* __restrict__ H, float* __restrict__ O, int Etot)
{
    constexpr int CPE = OUT / 4;
    long long g = (long long)blockIdx.x * 256 + threadIdx.x;
    int e  = (int)(g / CPE);
    if (e >= Etot) return;
    int ch = (int)(g % CPE);
    int s = src[e], d = dst[e];
    float al = Wg[e] / (Ssum[d] + 1e-16f);
    const float4 hv = *(const float4*)(H + (size_t)s * OUT + ch * 4);
    float* op = O + (size_t)d * OUT + ch * 4;
    atomicAdd(op + 0, al * hv.x);
    atomicAdd(op + 1, al * hv.y);
    atomicAdd(op + 2, al * hv.z);
    atomicAdd(op + 3, al * hv.w);
}

// ---------------------------------------------------------------------------
// Post-processing: bias+relu, batch norm, final diff
// ---------------------------------------------------------------------------
template<int OUT>
__global__ __launch_bounds__(256)
void bias_relu(const float* __restrict__ O, const float* __restrict__ b,
               float* __restrict__ T, int total)
{
    int i = blockIdx.x * 256 + threadIdx.x;
    if (i >= total) return;
    float t = O[i] + b[i & (OUT - 1)];
    T[i] = t > 0.f ? t : 0.f;
}

template<int OUT>
__global__ __launch_bounds__(256)
void bn_stats(const float* __restrict__ T, float* __restrict__ gsum,
              float* __restrict__ gsq, int total)
{
    __shared__ float ls[OUT], lq[OUT];
    int tid = threadIdx.x;
    if (tid < OUT) { ls[tid] = 0.f; lq[tid] = 0.f; }
    __syncthreads();
    float s = 0.f, q = 0.f;
    int stride = gridDim.x * 256;            // multiple of OUT -> col invariant
    for (long long i = (long long)blockIdx.x * 256 + tid; i < total; i += stride) {
        float v = T[i]; s += v; q += v * v;
    }
    int col = (blockIdx.x * 256 + tid) & (OUT - 1);
    atomicAdd(&ls[col], s);                  // ds_add_f32
    atomicAdd(&lq[col], q);
    __syncthreads();
    if (tid < OUT) { atomicAdd(&gsum[tid], ls[tid]); atomicAdd(&gsq[tid], lq[tid]); }
}

template<int OUT>
__global__ __launch_bounds__(256)
void bn_apply(float* __restrict__ T, const float* __restrict__ gsum,
              const float* __restrict__ gsq, const float* __restrict__ gam,
              const float* __restrict__ bet, int total, float invN)
{
    int i = blockIdx.x * 256 + threadIdx.x;
    if (i >= total) return;
    int col = i & (OUT - 1);
    float mu  = gsum[col] * invN;
    float var = gsq[col] * invN - mu * mu;
    T[i] = gam[col] * (T[i] - mu) * rsqrtf(var + 1e-5f) + bet[col];
}

__global__ __launch_bounds__(256)
void final_kernel(const float* __restrict__ O, const float* __restrict__ b,
                  const float* __restrict__ X, float* __restrict__ out, int total)
{
    int i = blockIdx.x * 256 + threadIdx.x;
    if (i >= total) return;
    float xh = O[i] + b[i & 127];
    xh = xh > 0.f ? xh : 0.f;
    out[i]         = fabsf(X[i] - xh);       // diff
    out[total + i] = xh;                     // x_hat
}

// ---------------------------------------------------------------------------
// Host-side layer driver
// ---------------------------------------------------------------------------
template<int IN, int OUT>
static void gat_core(const float* Xin, const float* W, const float* as_,
                     const float* ad_, const int* src, const int* dst, int Etot,
                     float* H, float* AS, float* AD, float* Mx, float* Ss,
                     float* Wg, float* O, hipStream_t stream)
{
    const int N = NNODES;
    gemm_attn<IN, OUT><<<(N + 127) / 128, 256, 0, stream>>>(Xin, W, as_, ad_, H, AS, AD, N);
    fill_kernel<<<(N + 255) / 256, 256, 0, stream>>>(Mx, -3.0e38f, N);
    fill_kernel<<<(N + 255) / 256, 256, 0, stream>>>(Ss, 0.f, N);
    fill_kernel<<<(N * OUT + 255) / 256, 256, 0, stream>>>(O, 0.f, N * OUT);
    edge_max<<<(Etot + 255) / 256, 256, 0, stream>>>(src, dst, AS, AD, Mx, Etot);
    edge_exp<<<(Etot + 255) / 256, 256, 0, stream>>>(src, dst, AS, AD, Mx, Ss, Wg, Etot);
    long long tw = (long long)Etot * (OUT / 4);
    edge_scatter<OUT><<<(int)((tw + 255) / 256), 256, 0, stream>>>(src, dst, Wg, Ss, H, O, Etot);
}

template<int OUT>
static void run_bn(float* T, const float* g, const float* be,
                   float* colsum, float* colsq, hipStream_t stream)
{
    const int total = NNODES * OUT;
    fill_kernel<<<1, 256, 0, stream>>>(colsum, 0.f, OUT);
    fill_kernel<<<1, 256, 0, stream>>>(colsq, 0.f, OUT);
    bn_stats<OUT><<<256, 256, 0, stream>>>(T, colsum, colsq, total);
    bn_apply<OUT><<<(total + 255) / 256, 256, 0, stream>>>(T, colsum, colsq, g, be,
                                                           total, 1.0f / NNODES);
}

extern "C" void kernel_launch(void* const* d_in, const int* in_sizes, int n_in,
                              void* d_out, int out_size, void* d_ws, size_t ws_size,
                              hipStream_t stream)
{
    (void)n_in; (void)out_size; (void)ws_size;
    const float*     x   = (const float*)d_in[0];
    const long long* ei  = (const long long*)d_in[1];   // jnp.int64 edge_index
    const float* W1 = (const float*)d_in[2];
    const float* a1s = (const float*)d_in[3];
    const float* a1d = (const float*)d_in[4];
    const float* b1  = (const float*)d_in[5];
    const float* g1  = (const float*)d_in[6];
    const float* be1 = (const float*)d_in[7];
    const float* W2 = (const float*)d_in[8];
    const float* a2s = (const float*)d_in[9];
    const float* a2d = (const float*)d_in[10];
    const float* b2  = (const float*)d_in[11];
    const float* g2  = (const float*)d_in[12];
    const float* be2 = (const float*)d_in[13];
    const float* W3 = (const float*)d_in[14];
    const float* a3s = (const float*)d_in[15];
    const float* a3d = (const float*)d_in[16];
    const float* b3  = (const float*)d_in[17];
    const float* g3  = (const float*)d_in[18];
    const float* be3 = (const float*)d_in[19];
    const float* W4 = (const float*)d_in[20];
    const float* a4s = (const float*)d_in[21];
    const float* a4d = (const float*)d_in[22];
    const float* b4  = (const float*)d_in[23];

    const int N = NNODES;
    const int E = in_sizes[1] / 2;
    const int Etot = E + N;

    // workspace carve (256B aligned)
    char* cur = (char*)d_ws;
    auto carve = [&](size_t bytes) -> void* {
        void* r = (void*)cur;
        cur += (bytes + 255) & ~(size_t)255;
        return r;
    };
    int*   src = (int*)carve((size_t)Etot * 4);
    int*   dst = (int*)carve((size_t)Etot * 4);
    float* Wg  = (float*)carve((size_t)Etot * 4);
    float* AS  = (float*)carve((size_t)N * 4);
    float* AD  = (float*)carve((size_t)N * 4);
    float* Mx  = (float*)carve((size_t)N * 4);
    float* Ss  = (float*)carve((size_t)N * 4);
    float* H   = (float*)carve((size_t)N * 128 * 4);
    float* O   = (float*)carve((size_t)N * 128 * 4);
    float* F1  = (float*)carve((size_t)N * 128 * 4);
    float* F2  = (float*)carve((size_t)N * 128 * 4);
    float* colsum = (float*)carve(128 * 4);
    float* colsq  = (float*)carve(128 * 4);

    build_edges<<<(Etot + 255) / 256, 256, 0, stream>>>(ei, src, dst, E, Etot);

    // Layer 1: 128 -> 64, relu + bn
    gat_core<128, 64>(x, W1, a1s, a1d, src, dst, Etot, H, AS, AD, Mx, Ss, Wg, O, stream);
    bias_relu<64><<<(N * 64 + 255) / 256, 256, 0, stream>>>(O, b1, F1, N * 64);
    run_bn<64>(F1, g1, be1, colsum, colsq, stream);

    // Layer 2: 64 -> 64, relu + bn
    gat_core<64, 64>(F1, W2, a2s, a2d, src, dst, Etot, H, AS, AD, Mx, Ss, Wg, O, stream);
    bias_relu<64><<<(N * 64 + 255) / 256, 256, 0, stream>>>(O, b2, F2, N * 64);
    run_bn<64>(F2, g2, be2, colsum, colsq, stream);

    // Layer 3: 64 -> 64, relu + bn
    gat_core<64, 64>(F2, W3, a3s, a3d, src, dst, Etot, H, AS, AD, Mx, Ss, Wg, O, stream);
    bias_relu<64><<<(N * 64 + 255) / 256, 256, 0, stream>>>(O, b3, F1, N * 64);
    run_bn<64>(F1, g3, be3, colsum, colsq, stream);

    // Layer 4: 64 -> 128, relu, diff
    gat_core<64, 128>(F1, W4, a4s, a4d, src, dst, Etot, H, AS, AD, Mx, Ss, Wg, O, stream);
    final_kernel<<<(N * 128 + 255) / 256, 256, 0, stream>>>(O, b4, x, (float*)d_out, N * 128);
}
